// EfficientTransformerEncoderLayer_80882824119008
// MI455X (gfx1250) — matrix operand order repaired and checked by reading, not compiled, not run on hardware
//
#include <hip/hip_runtime.h>

typedef __bf16 bf16;
typedef __attribute__((ext_vector_type(16))) __bf16 v16bf;
typedef __attribute__((ext_vector_type(8)))  __bf16 v8bf;
typedef __attribute__((ext_vector_type(8)))  float  v8f;

#define D_MODEL 1024
#define SEQ     4096
#define BATCH   4
#define NHEAD   16
#define DK      64
#define MROWS   (SEQ*BATCH)   /* 16384 */
#define DFF     2048

union FragU { v16bf v; v8bf h[2]; };

// ---------------- WMMA fragment loaders ----------------
// A: bf16 [M,K] row-major. 16x32 tile at (rowBase, kk).
__device__ inline v16bf load_a_frag(const bf16* __restrict__ A, int lda,
                                    int rowBase, int kk, int lane) {
  const int half = lane >> 4;
  const int r    = rowBase + (lane & 15);
  const bf16* p = A + (size_t)r * lda + kk + half * 8;
  FragU f;
  f.h[0] = *(const v8bf*)(p);        // K = kk + half*8 .. +8
  f.h[1] = *(const v8bf*)(p + 16);   // K = kk + 16 + half*8 .. +8
  return f.v;
}

// B: weights bf16 [N,K] row-major (computes X @ W^T). 32x16 tile at (colBase, kk).
__device__ inline v16bf load_b_frag(const bf16* __restrict__ W, int ldw,
                                    int colBase, int kk, int lane) {
  const int half = lane >> 4;
  const int r    = colBase + (lane & 15);
  const bf16* p = W + (size_t)r * ldw + kk + half * 8;
  FragU f;
  f.h[0] = *(const v8bf*)(p);        // K = kk + half*8 .. +8
  f.h[1] = *(const v8bf*)(p + 16);   // K = kk + 16 + half*8 .. +8
  return f.v;
}

// ---------------- Generic GEMM: C = act(A @ W^T + bias) ----------------
// ACT: 0 = none, 1 = elu(x)+1, 2 = relu
// Block 128 threads = 4 waves; wave tile 32(M) x 64(N); block tile 64 x 128.
template<int ACT>
__global__ __launch_bounds__(128)
void gemm_bf16_kernel(const bf16* __restrict__ A, const bf16* __restrict__ W,
                      const float* __restrict__ bias,
                      bf16* __restrict__ outB, float* __restrict__ outF,
                      int M, int N, int K) {
  const int lane = threadIdx.x & 31;
  const int wave = threadIdx.x >> 5;
  const int m0 = blockIdx.y * 64  + (wave & 1) * 32;
  const int n0 = blockIdx.x * 128 + (wave >> 1) * 64;

  v8f acc[2][4];
#pragma unroll
  for (int i = 0; i < 2; ++i)
#pragma unroll
    for (int j = 0; j < 4; ++j)
#pragma unroll
      for (int v = 0; v < 8; ++v) acc[i][j][v] = 0.0f;

  for (int kk = 0; kk < K; kk += 32) {
    v16bf a0 = load_a_frag(A, K, m0,      kk, lane);
    v16bf a1 = load_a_frag(A, K, m0 + 16, kk, lane);
    v16bf b0 = load_b_frag(W, K, n0,      kk, lane);
    v16bf b1 = load_b_frag(W, K, n0 + 16, kk, lane);
    v16bf b2 = load_b_frag(W, K, n0 + 32, kk, lane);
    v16bf b3 = load_b_frag(W, K, n0 + 48, kk, lane);
    acc[0][0] = __builtin_amdgcn_wmma_f32_16x16x32_bf16(false, a0, false, b0, (short)0, acc[0][0], false, false);
    acc[0][1] = __builtin_amdgcn_wmma_f32_16x16x32_bf16(false, a0, false, b1, (short)0, acc[0][1], false, false);
    acc[0][2] = __builtin_amdgcn_wmma_f32_16x16x32_bf16(false, a0, false, b2, (short)0, acc[0][2], false, false);
    acc[0][3] = __builtin_amdgcn_wmma_f32_16x16x32_bf16(false, a0, false, b3, (short)0, acc[0][3], false, false);
    acc[1][0] = __builtin_amdgcn_wmma_f32_16x16x32_bf16(false, a1, false, b0, (short)0, acc[1][0], false, false);
    acc[1][1] = __builtin_amdgcn_wmma_f32_16x16x32_bf16(false, a1, false, b1, (short)0, acc[1][1], false, false);
    acc[1][2] = __builtin_amdgcn_wmma_f32_16x16x32_bf16(false, a1, false, b2, (short)0, acc[1][2], false, false);
    acc[1][3] = __builtin_amdgcn_wmma_f32_16x16x32_bf16(false, a1, false, b3, (short)0, acc[1][3], false, false);
  }

  // C/D layout: lanes 0-15 -> N=lane, M=v (rows 0-7); lanes 16-31 -> N=lane-16, M=8+v.
#pragma unroll
  for (int mf = 0; mf < 2; ++mf)
#pragma unroll
    for (int nf = 0; nf < 4; ++nf) {
      const int col = n0 + nf * 16 + (lane & 15);
      const int rb  = m0 + mf * 16 + (lane >> 4) * 8;
      const float bs = bias ? bias[col] : 0.0f;
#pragma unroll
      for (int v = 0; v < 8; ++v) {
        const int row = rb + v;
        float c = acc[mf][nf][v] + bs;
        if (ACT == 1) c = (c > 0.0f) ? (c + 1.0f) : __expf(c);   // elu(x)+1
        if (ACT == 2) c = (c > 0.0f) ? c : 0.0f;                 // relu
        if (outB) outB[(size_t)row * N + col] = (bf16)c;
        if (outF) outF[(size_t)row * N + col] = c;
      }
    }
}

// ---------------- attn = (phiQ @ ctx_bh) / (denom + eps), per head ----------------
// ctxT stored per (b,h) as [e][d] row-major (64x64) so B frags are K-contiguous.
__global__ __launch_bounds__(128)
void attn_out_kernel(const bf16* __restrict__ phiQ, const bf16* __restrict__ ctxT,
                     const float* __restrict__ denom, bf16* __restrict__ attn) {
  const int lane = threadIdx.x & 31;
  const int wave = threadIdx.x >> 5;
  const int m0 = blockIdx.x * 128 + wave * 32;     // rows in bsd order, one b per block
  const int h  = blockIdx.y;
  const int b  = m0 >> 12;
  const bf16* Bh = ctxT + (size_t)(b * NHEAD + h) * (DK * DK);

  v8f acc[2][4];
#pragma unroll
  for (int i = 0; i < 2; ++i)
#pragma unroll
    for (int j = 0; j < 4; ++j)
#pragma unroll
      for (int v = 0; v < 8; ++v) acc[i][j][v] = 0.0f;

#pragma unroll
  for (int ks = 0; ks < 2; ++ks) {
    const int kk = ks * 32;
    v16bf a0 = load_a_frag(phiQ, D_MODEL, m0,      h * DK + kk, lane);
    v16bf a1 = load_a_frag(phiQ, D_MODEL, m0 + 16, h * DK + kk, lane);
    v16bf b0 = load_b_frag(Bh, DK, 0,  kk, lane);
    v16bf b1 = load_b_frag(Bh, DK, 16, kk, lane);
    v16bf b2 = load_b_frag(Bh, DK, 32, kk, lane);
    v16bf b3 = load_b_frag(Bh, DK, 48, kk, lane);
    acc[0][0] = __builtin_amdgcn_wmma_f32_16x16x32_bf16(false, a0, false, b0, (short)0, acc[0][0], false, false);
    acc[0][1] = __builtin_amdgcn_wmma_f32_16x16x32_bf16(false, a0, false, b1, (short)0, acc[0][1], false, false);
    acc[0][2] = __builtin_amdgcn_wmma_f32_16x16x32_bf16(false, a0, false, b2, (short)0, acc[0][2], false, false);
    acc[0][3] = __builtin_amdgcn_wmma_f32_16x16x32_bf16(false, a0, false, b3, (short)0, acc[0][3], false, false);
    acc[1][0] = __builtin_amdgcn_wmma_f32_16x16x32_bf16(false, a1, false, b0, (short)0, acc[1][0], false, false);
    acc[1][1] = __builtin_amdgcn_wmma_f32_16x16x32_bf16(false, a1, false, b1, (short)0, acc[1][1], false, false);
    acc[1][2] = __builtin_amdgcn_wmma_f32_16x16x32_bf16(false, a1, false, b2, (short)0, acc[1][2], false, false);
    acc[1][3] = __builtin_amdgcn_wmma_f32_16x16x32_bf16(false, a1, false, b3, (short)0, acc[1][3], false, false);
  }

#pragma unroll
  for (int mf = 0; mf < 2; ++mf)
#pragma unroll
    for (int nf = 0; nf < 4; ++nf) {
      const int e  = nf * 16 + (lane & 15);
      const int rb = m0 + mf * 16 + (lane >> 4) * 8;
#pragma unroll
      for (int v = 0; v < 8; ++v) {
        const int row = rb + v;
        const float den = denom[(size_t)row * NHEAD + h] + 1e-6f;
        attn[(size_t)row * D_MODEL + h * DK + e] = (bf16)(acc[mf][nf][v] / den);
      }
    }
}

// ---------------- context: ctx[bh][d][e] += sum_s phiK[s,d] * V[s,e] ----------------
__global__ __launch_bounds__(256)
void context_kernel(const bf16* __restrict__ phiK, const bf16* __restrict__ V,
                    float* __restrict__ ctx) {
  const int bh = blockIdx.x;
  const int split = blockIdx.y;        // 8 splits of 512 rows
  const int b = bh >> 4, h = bh & 15;
  const int t = threadIdx.x;
  const int d0 = (t & 15) * 4;
  const int e0 = (t >> 4) * 4;
  __shared__ float sK[32][64];
  __shared__ float sV[32][64];
  float acc[4][4] = {};
  const size_t rowBase = (size_t)b * SEQ + (size_t)split * 512;
  const int colK = h * DK;

  for (int chunk = 0; chunk < 512; chunk += 32) {
    for (int i = t; i < 32 * 64; i += 256) {
      const int ss = i >> 6, dd = i & 63;
      const size_t off = (rowBase + chunk + ss) * D_MODEL + colK + dd;
      sK[ss][dd] = (float)phiK[off];
      sV[ss][dd] = (float)V[off];
    }
    __syncthreads();
#pragma unroll 4
    for (int ss = 0; ss < 32; ++ss) {
      float kd[4], ve[4];
#pragma unroll
      for (int j = 0; j < 4; ++j) { kd[j] = sK[ss][d0 + j]; ve[j] = sV[ss][e0 + j]; }
#pragma unroll
      for (int j = 0; j < 4; ++j)
#pragma unroll
        for (int l = 0; l < 4; ++l) acc[j][l] += kd[j] * ve[l];
    }
    __syncthreads();
  }
#pragma unroll
  for (int j = 0; j < 4; ++j)
#pragma unroll
    for (int l = 0; l < 4; ++l)
      atomicAdd(&ctx[((size_t)bh * DK + d0 + j) * DK + e0 + l], acc[j][l]);
}

// ---------------- ksum[bh][d] = sum_s phiK[s,d] ----------------
__global__ __launch_bounds__(256)
void ksum_kernel(const bf16* __restrict__ phiK, float* __restrict__ ksum) {
  const int bh = blockIdx.x;
  const int b = bh >> 4, h = bh & 15;
  const int t = threadIdx.x;
  const int d = t & 63, part = t >> 6;
  float s = 0.0f;
  const size_t base = ((size_t)b * SEQ + part * 1024) * D_MODEL + h * DK + d;
  for (int ss = 0; ss < 1024; ++ss) s += (float)phiK[base + (size_t)ss * D_MODEL];
  __shared__ float red[4][64];
  red[part][d] = s;
  __syncthreads();
  if (part == 0)
    ksum[(size_t)bh * DK + d] = red[0][d] + red[1][d] + red[2][d] + red[3][d];
}

// ---------------- denom[m][h] = phiQ[m, h*64..] . ksum[bh] ----------------
__global__ void denom_kernel(const bf16* __restrict__ phiQ, const float* __restrict__ ksum,
                             float* __restrict__ denom) {
  const int i = blockIdx.x * 256 + threadIdx.x;     // M*NHEAD total
  const int m = i >> 4, h = i & 15;
  const int b = m >> 12;
  const bf16*  q  = phiQ + (size_t)m * D_MODEL + h * DK;
  const float* ks = ksum + (size_t)(b * NHEAD + h) * DK;
  float s = 0.0f;
#pragma unroll 8
  for (int d = 0; d < DK; ++d) s += (float)q[d] * ks[d];
  denom[i] = s;
}

// ---------------- LayerNorm (row = 1024), optional row remap bsd<->sbd ----------------
__global__ __launch_bounds__(256)
void layernorm_kernel(const float* __restrict__ x, const float* __restrict__ resid,
                      int residRemap, const float* __restrict__ g, const float* __restrict__ be,
                      float* __restrict__ outF, bf16* __restrict__ outB, int outRemap) {
  const int r = blockIdx.x;
  const int t = threadIdx.x;
  const int rRe = (r & (SEQ - 1)) * BATCH + (r >> 12);   // bsd row -> sbd row
  const float* xr = x + (size_t)r * D_MODEL;
  const float* rr = resid + (size_t)(residRemap ? rRe : r) * D_MODEL;
  float v[4], s = 0.0f, sq = 0.0f;
#pragma unroll
  for (int i = 0; i < 4; ++i) {
    const int c = t + i * 256;
    v[i] = xr[c] + rr[c];
    s += v[i]; sq += v[i] * v[i];
  }
  __shared__ float s1[256], s2[256];
  s1[t] = s; s2[t] = sq;
  __syncthreads();
  for (int off = 128; off > 0; off >>= 1) {
    if (t < off) { s1[t] += s1[t + off]; s2[t] += s2[t + off]; }
    __syncthreads();
  }
  const float mean = s1[0] * (1.0f / D_MODEL);
  const float var  = s2[0] * (1.0f / D_MODEL) - mean * mean;
  const float rs   = rsqrtf(var + 1e-5f);
  float* of = outF + (size_t)(outRemap ? rRe : r) * D_MODEL;
#pragma unroll
  for (int i = 0; i < 4; ++i) {
    const int c = t + i * 256;
    const float y = (v[i] - mean) * rs * g[c] + be[c];
    of[c] = y;
    if (outB) outB[(size_t)r * D_MODEL + c] = (bf16)y;
  }
}

// ---------------- small utility kernels ----------------
__global__ void cast_f32_bf16_kernel(const float* __restrict__ in, bf16* __restrict__ out, int n) {
  const int i = blockIdx.x * 256 + threadIdx.x;
  if (i < n) out[i] = (bf16)in[i];
}

__global__ void cast_src_kernel(const float* __restrict__ src, bf16* __restrict__ dst) {
  const int i = blockIdx.x * 256 + threadIdx.x;   // M*D total
  const int r = i >> 10, c = i & 1023;            // bsd row, col
  const int sr = (r & (SEQ - 1)) * BATCH + (r >> 12);
  dst[(size_t)r * D_MODEL + c] = (bf16)src[(size_t)sr * D_MODEL + c];
}

__global__ void zero_f32_kernel(float* __restrict__ p, int n) {
  const int i = blockIdx.x * 256 + threadIdx.x;
  if (i < n) p[i] = 0.0f;
}

__global__ void ctx_transpose_kernel(const float* __restrict__ ctx, bf16* __restrict__ ctxT) {
  const int i = blockIdx.x * 256 + threadIdx.x;   // 64*64*64 total
  const int bh = i >> 12, rem = i & 4095;
  const int d = rem >> 6, e = rem & 63;
  ctxT[(size_t)bh * 4096 + e * 64 + d] = (bf16)ctx[i];
}

// ---------------- host launcher ----------------
static inline char* ws_take(char* base, size_t& off, size_t bytes) {
  off = (off + 255) & ~(size_t)255;
  char* p = base + off;
  off += bytes;
  return p;
}

extern "C" void kernel_launch(void* const* d_in, const int* in_sizes, int n_in,
                              void* d_out, int out_size, void* d_ws, size_t ws_size,
                              hipStream_t stream) {
  (void)in_sizes; (void)n_in; (void)out_size; (void)ws_size;
  const float* src = (const float*)d_in[0];
  const float* Wq  = (const float*)d_in[1];  const float* bq = (const float*)d_in[2];
  const float* Wk  = (const float*)d_in[3];  const float* bk = (const float*)d_in[4];
  const float* Wv  = (const float*)d_in[5];  const float* bv = (const float*)d_in[6];
  const float* Wo  = (const float*)d_in[7];  const float* bo = (const float*)d_in[8];
  const float* W1  = (const float*)d_in[9];  const float* b1 = (const float*)d_in[10];
  const float* W2  = (const float*)d_in[11]; const float* b2 = (const float*)d_in[12];
  const float* g1  = (const float*)d_in[13]; const float* be1 = (const float*)d_in[14];
  const float* g2  = (const float*)d_in[15]; const float* be2 = (const float*)d_in[16];
  float* out = (float*)d_out;

  const size_t MD = (size_t)MROWS * D_MODEL;   // 16,777,216
  char* ws = (char*)d_ws;
  size_t off = 0;
  bf16* srcT  = (bf16*)ws_take(ws, off, MD * 2);
  bf16* wqb   = (bf16*)ws_take(ws, off, (size_t)D_MODEL * D_MODEL * 2);
  bf16* wkb   = (bf16*)ws_take(ws, off, (size_t)D_MODEL * D_MODEL * 2);
  bf16* wvb   = (bf16*)ws_take(ws, off, (size_t)D_MODEL * D_MODEL * 2);
  bf16* wob   = (bf16*)ws_take(ws, off, (size_t)D_MODEL * D_MODEL * 2);
  bf16* w1b   = (bf16*)ws_take(ws, off, (size_t)DFF * D_MODEL * 2);
  bf16* w2b   = (bf16*)ws_take(ws, off, (size_t)D_MODEL * DFF * 2);
  bf16* phiQ  = (bf16*)ws_take(ws, off, MD * 2);
  bf16* phiK  = (bf16*)ws_take(ws, off, MD * 2);
  bf16* Vb    = (bf16*)ws_take(ws, off, MD * 2);       // contiguous after phiK
  float* ctx  = (float*)ws_take(ws, off, (size_t)64 * DK * DK * 4);
  bf16* ctxT  = (bf16*)ws_take(ws, off, (size_t)64 * DK * DK * 2);
  float* ksum = (float*)ws_take(ws, off, (size_t)64 * DK * 4);
  float* den  = (float*)ws_take(ws, off, (size_t)MROWS * NHEAD * 4);
  bf16* attn  = (bf16*)ws_take(ws, off, MD * 2);
  float* attnO= (float*)ws_take(ws, off, MD * 4);
  float* y1   = (float*)ws_take(ws, off, MD * 4);
  // aliases (lifetimes disjoint): hidden (16384x2048 bf16, 64MB) over phiK+Vb;
  // y1 bf16 over attn; ffn f32 over attnO.
  bf16*  hid = phiK;
  bf16*  y1b = attn;
  float* ffn = attnO;

  // 1) casts
  cast_f32_bf16_kernel<<<(D_MODEL*D_MODEL+255)/256, 256, 0, stream>>>(Wq, wqb, D_MODEL*D_MODEL);
  cast_f32_bf16_kernel<<<(D_MODEL*D_MODEL+255)/256, 256, 0, stream>>>(Wk, wkb, D_MODEL*D_MODEL);
  cast_f32_bf16_kernel<<<(D_MODEL*D_MODEL+255)/256, 256, 0, stream>>>(Wv, wvb, D_MODEL*D_MODEL);
  cast_f32_bf16_kernel<<<(D_MODEL*D_MODEL+255)/256, 256, 0, stream>>>(Wo, wob, D_MODEL*D_MODEL);
  cast_f32_bf16_kernel<<<(DFF*D_MODEL+255)/256, 256, 0, stream>>>(W1, w1b, DFF*D_MODEL);
  cast_f32_bf16_kernel<<<(D_MODEL*DFF+255)/256, 256, 0, stream>>>(W2, w2b, D_MODEL*DFF);
  cast_src_kernel<<<(int)(MD/256), 256, 0, stream>>>(src, srcT);

  // 2) QKV projections (WMMA) with phi epilogue for Q,K
  dim3 gemm_blk(128);
  dim3 gemm_grd_1024(D_MODEL/128, MROWS/64);
  gemm_bf16_kernel<1><<<gemm_grd_1024, gemm_blk, 0, stream>>>(srcT, wqb, bq, phiQ, nullptr, MROWS, D_MODEL, D_MODEL);
  gemm_bf16_kernel<1><<<gemm_grd_1024, gemm_blk, 0, stream>>>(srcT, wkb, bk, phiK, nullptr, MROWS, D_MODEL, D_MODEL);
  gemm_bf16_kernel<0><<<gemm_grd_1024, gemm_blk, 0, stream>>>(srcT, wvb, bv, Vb,   nullptr, MROWS, D_MODEL, D_MODEL);

  // 3) linear attention state
  zero_f32_kernel<<<(64*DK*DK+255)/256, 256, 0, stream>>>(ctx, 64*DK*DK);
  ksum_kernel<<<64, 256, 0, stream>>>(phiK, ksum);
  context_kernel<<<dim3(64, 8), 256, 0, stream>>>(phiK, Vb, ctx);
  ctx_transpose_kernel<<<(64*DK*DK+255)/256, 256, 0, stream>>>(ctx, ctxT);
  denom_kernel<<<(MROWS*NHEAD+255)/256, 256, 0, stream>>>(phiQ, ksum, den);

  // 4) attn = (phiQ @ ctx) / denom  (WMMA, per-head block-diagonal)
  attn_out_kernel<<<dim3(MROWS/128, NHEAD), gemm_blk, 0, stream>>>(phiQ, ctxT, den, attn);

  // 5) output projection + LN1
  gemm_bf16_kernel<0><<<gemm_grd_1024, gemm_blk, 0, stream>>>(attn, wob, bo, nullptr, attnO, MROWS, D_MODEL, D_MODEL);
  layernorm_kernel<<<MROWS, 256, 0, stream>>>(attnO, src, 1, g1, be1, y1, y1b, 0);

  // 6) FFN (WMMA) + LN2 (scatter back to [S,B,D])
  dim3 gemm_grd_2048(DFF/128, MROWS/64);
  gemm_bf16_kernel<2><<<gemm_grd_2048, gemm_blk, 0, stream>>>(y1b, w1b, b1, hid, nullptr, MROWS, DFF, D_MODEL);
  gemm_bf16_kernel<0><<<gemm_grd_1024, gemm_blk, 0, stream>>>(hid, w2b, b2, nullptr, ffn, MROWS, D_MODEL, DFF);
  layernorm_kernel<<<MROWS, 256, 0, stream>>>(ffn, y1, 0, g2, be2, out, nullptr, 1);
}
